// QAggregateStgcn_18760417148965
// MI455X (gfx1250) — 8-bit matrix operand order repaired and checked by reading, compiled, not run on hardware
//
#include <hip/hip_runtime.h>

typedef __attribute__((ext_vector_type(8))) int v8i;

#define N_   4096
#define A_   3
#define C_   64
#define V_   25
#define G_   9
#define K_   (A_ * V_)
#define GV_  (G_ * V_)
#define ACT_SCALE 0.05f
#define ROWB 136

__device__ __forceinline__ float div_act_scale(float x) {
    constexpr float c = ACT_SCALE;
    constexpr float r = 1.0f / ACT_SCALE;
    float q0 = x * r;
    float e  = __builtin_fmaf(-c, q0, x);
    return __builtin_fmaf(e, r, q0);
}

__device__ __forceinline__ float q128f(float x) {
    float q = __builtin_rintf(div_act_scale(x) + 128.0f);
    return fminf(fmaxf(q, 0.0f), 255.0f);
}

__global__ void qstgcn_prep(const float* __restrict__ w_matmul,
                            const float* __restrict__ w_sum,
                            const float* __restrict__ b_sum,
                            int* __restrict__ Bpk,
                            float* __restrict__ cst) {
    __shared__ float red[256];
    int t = threadIdx.x;
    float m = 0.0f;
    for (int i = t; i < V_ * K_; i += 256) m = fmaxf(m, fabsf(w_matmul[i]));
    red[t] = m;
    __syncthreads();
    for (int s = 128; s > 0; s >>= 1) {
        if (t < s) red[t] = fmaxf(red[t], red[t + s]);
        __syncthreads();
    }
    float sw = fmaxf(red[0] / 127.0f, 1e-8f);

    for (int idx = t; idx < 2 * 2 * 8 * 32; idx += 256) {
        int lane = idx & 31;
        int vg   = (idx >> 5) & 7;
        int nt   = (idx >> 8) & 1;
        int kt   = (idx >> 9) & 1;
        int col  = (lane & 15) + nt * 16;
        int kbase = kt * 64 + (vg >> 1) * 16 + ((lane >> 4) & 1) * 8 + (vg & 1) * 4;
        unsigned pack = 0u;
        for (int b = 0; b < 4; ++b) {
            int k = kbase + b;
            int q = 0;
            if (k < K_ && col < V_) {
                float qf = __builtin_rintf(w_matmul[col * K_ + k] / sw);
                qf = fminf(fmaxf(qf, -128.0f), 127.0f);
                q = (int)qf;
            }
            pack |= ((unsigned)(q & 0xFF)) << (8 * b);
        }
        Bpk[idx] = (int)pack;
    }

    if (t == 0) {
        float mw = 0.0f;
        for (int g = 0; g < G_; ++g) mw = fmaxf(mw, fabsf(w_sum[g]));
        float ss = fmaxf(mw / 127.0f, 1e-8f);
        float sum = 0.0f;
        for (int g = 0; g < G_; ++g) {
            float qf = fminf(fmaxf(__builtin_rintf(w_sum[g] / ss), -128.0f), 127.0f);
            float wq = qf * ss;
            cst[g] = wq * ACT_SCALE;
            sum += wq;
        }
        cst[9]  = b_sum[0] - 128.0f * ACT_SCALE * sum;
        cst[10] = ACT_SCALE * sw;
    }
}

__global__ void __launch_bounds__(256) qstgcn_main(
        const float* __restrict__ x,
        const float* __restrict__ fifo,
        const float* __restrict__ b_matmul,
        const int*   __restrict__ Bpk,
        const float* __restrict__ cst,
        float* __restrict__ out) {
    __shared__ char lds[8][16 * ROWB];
    int lane = threadIdx.x & 31;
    int wib  = threadIdx.x >> 5;
    int wave = blockIdx.x * 8 + wib;
    int n    = wave >> 2;
    int c0   = (wave & 3) << 4;
    char* xq = lds[wib];

    {
        int4* p = (int4*)xq;
        for (int i = lane; i < (16 * ROWB) / 16; i += 32) p[i] = make_int4(0, 0, 0, 0);
    }
    const float* xb = x + (size_t)n * (A_ * C_ * V_) + (size_t)c0 * V_;
    for (int e = lane; e < A_ * 16 * V_; e += 32) {
        int a   = e / (16 * V_);
        int rem = e - a * (16 * V_);
        int cl  = rem / V_;
        int v   = rem - cl * V_;
        float xv = xb[a * (C_ * V_) + cl * V_ + v];
        xq[cl * ROWB + a * V_ + v] = (char)((int)q128f(xv) - 128);
    }
    __asm__ volatile("s_wait_dscnt 0" ::: "memory");

    int row = lane & 15;
    int hb  = (lane >> 4) << 3;
    const char* xr = xq + row * ROWB + hb;
    constexpr int koff[8] = {0, 4, 16, 20, 32, 36, 48, 52};
    v8i a0, a1;
#pragma unroll
    for (int i = 0; i < 8; ++i) {
        a0[i] = *(const int*)(xr + koff[i]);
        a1[i] = *(const int*)(xr + 64 + koff[i]);
    }
    v8i b00, b01, b10, b11;
#pragma unroll
    for (int i = 0; i < 8; ++i) {
        b00[i] = Bpk[(0 * 8 + i) * 32 + lane];
        b01[i] = Bpk[(1 * 8 + i) * 32 + lane];
        b10[i] = Bpk[(2 * 8 + i) * 32 + lane];
        b11[i] = Bpk[(3 * 8 + i) * 32 + lane];
    }

    v8i acc0 = {}, acc1 = {};
    acc0 = __builtin_amdgcn_wmma_i32_16x16x64_iu8(true, a0, true, b00, acc0, false, false);
    acc0 = __builtin_amdgcn_wmma_i32_16x16x64_iu8(true, a1, true, b10, acc0, false, false);
    acc1 = __builtin_amdgcn_wmma_i32_16x16x64_iu8(true, a0, true, b01, acc1, false, false);
    acc1 = __builtin_amdgcn_wmma_i32_16x16x64_iu8(true, a1, true, b11, acc1, false, false);

    float wg[9];
#pragma unroll
    for (int g = 0; g < 9; ++g) wg[g] = cst[g];
    float bias   = cst[9];
    float yscale = cst[10];

    int c_base = c0 + ((lane >> 4) << 3);
    const float* fbase = fifo + (size_t)(n * C_ + c_base) * GV_ + (lane & 15);
    float*       obase = out  + (size_t)(n * C_ + c_base) * V_  + (lane & 15);
#pragma unroll
    for (int nt = 0; nt < 2; ++nt) {
        int v = (lane & 15) + nt * 16;
        bool ok = (v < V_);
        float bm = ok ? b_matmul[v] : 0.0f;
#pragma unroll
        for (int r = 0; r < 8; ++r) {
            int accv = nt ? acc1[r] : acc0[r];
            float y = __builtin_fmaf((float)accv, yscale, bm);
            float s = __builtin_fmaf(wg[0], q128f(y), bias);
#pragma unroll
            for (int g = 0; g < 8; ++g) {
                float f = __builtin_nontemporal_load(&fbase[nt * 16 + r * GV_ + g * V_]);
                s = __builtin_fmaf(wg[g + 1], q128f(f), s);
            }
            float o = __builtin_fmaf(q128f(s), ACT_SCALE, -128.0f * ACT_SCALE);
            if (ok) __builtin_nontemporal_store(o, &obase[nt * 16 + r * V_]);
        }
    }
}

extern "C" void kernel_launch(void* const* d_in, const int* in_sizes, int n_in,
                              void* d_out, int out_size, void* d_ws, size_t ws_size,
                              hipStream_t stream) {
    const float* x        = (const float*)d_in[0];
    const float* fifo     = (const float*)d_in[1];
    const float* w_matmul = (const float*)d_in[2];
    const float* b_matmul = (const float*)d_in[3];
    const float* w_sum    = (const float*)d_in[4];
    const float* b_sum    = (const float*)d_in[5];

    int*   Bpk = (int*)d_ws;
    float* cst = (float*)((char*)d_ws + 4096);

    qstgcn_prep<<<1, 256, 0, stream>>>(w_matmul, w_sum, b_sum, Bpk, cst);

    int blocks = (N_ * 4) / 8;
    qstgcn_main<<<blocks, 256, 0, stream>>>(x, fifo, b_matmul, Bpk, cst,
                                            (float*)d_out);
}
